// TwoMatMulModel_53953379172430
// MI455X (gfx1250) — matrix operand order repaired and checked by reading, not compiled, not run on hardware
//
#include <hip/hip_runtime.h>

#if __has_builtin(__builtin_amdgcn_global_load_async_to_lds_b128) && \
    __has_builtin(__builtin_amdgcn_s_wait_asynccnt)
#define USE_ASYNC 1
#else
#define USE_ASYNC 0
#endif

#if USE_ASYNC && __has_builtin(__builtin_amdgcn_tensor_load_to_lds) && \
    __has_builtin(__builtin_amdgcn_s_wait_tensorcnt)
#define USE_TDM 1
#else
#define USE_TDM 0
#endif

#define AS1 __attribute__((address_space(1)))
#define AS3 __attribute__((address_space(3)))

#define NDIM 4096
#define KSTEP 64
#define LSTRIDE 144  // 128B of K data + 16B pad: keeps 16B alignment, spreads banks

typedef __bf16 bf16;
typedef __attribute__((ext_vector_type(8)))  __bf16 bf16x8;
typedef __attribute__((ext_vector_type(16))) __bf16 bf16x16;
typedef __attribute__((ext_vector_type(8)))  float  f32x8;
typedef int          i32x4 __attribute__((vector_size(16)));
typedef unsigned int u32x4 __attribute__((vector_size(16)));
typedef int          i32x8 __attribute__((vector_size(32)));

// Exact E8M0 scale per MX spec: 2^(floor(log2(amax)) - 8), clamped; plus exact inverse.
__device__ inline void mx_scale(float amax, float& s, float& inv) {
  if (!(amax > 0.0f)) amax = 1.0f;
  int e  = (int)((__float_as_uint(amax) >> 23) & 0xFFu) - 127;
  int se = e - 8;
  se = se < -126 ? -126 : (se > 126 ? 126 : se);
  s   = __uint_as_float((unsigned)(se + 127) << 23);
  inv = __uint_as_float((unsigned)(127 - se) << 23);
}

// Software float -> e4m3 (RNE, saturate 448, denorm step 2^-9) -> float.
__device__ inline float e4m3_qdq(float y) {
  float a = __builtin_fabsf(y);
  a = fminf(a, 448.0f);
  float r;
  if (a < 0.015625f) {
    r = rintf(a * 512.0f) * 0.001953125f;
  } else {
    unsigned b = __float_as_uint(a);
    b = (b + 0x7FFFFu + ((b >> 20) & 1u)) & 0xFFF00000u;
    r = __uint_as_float(b);
  }
  return __builtin_copysignf(r, y);
}

// XOR-max across lanes via ds_swizzle (group-of-32: offset = xor<<10 | and=0x1f).
#define XORMAX(v, m)                                                            \
  (v) = fmaxf((v), __int_as_float(__builtin_amdgcn_ds_swizzle(                  \
                  __float_as_int(v), ((m) << 10) | 0x1f)))

// ---------------- Pass 1: MX quant/dequant, store bf16 (exact for MX values) -----
template <bool TRANSPOSE>
__global__ void mx_qdq_kernel(const float* __restrict__ in, bf16* __restrict__ out) {
  const int t = blockIdx.x * blockDim.x + threadIdx.x;
  const int row = TRANSPOSE ? (t & (NDIM - 1)) : (t >> 7);
  const int blk = TRANSPOSE ? (t >> 12) : (t & 127);
  const float* p = in + (size_t)row * NDIM + blk * 32;

  float v[32];
#pragma unroll
  for (int i = 0; i < 8; ++i) {
    float4 f = ((const float4*)p)[i];
    v[4 * i + 0] = f.x; v[4 * i + 1] = f.y; v[4 * i + 2] = f.z; v[4 * i + 3] = f.w;
  }
  float am = 0.0f;
#pragma unroll
  for (int i = 0; i < 32; ++i) am = fmaxf(am, __builtin_fabsf(v[i]));
  float s, inv;
  mx_scale(am, s, inv);
#pragma unroll
  for (int i = 0; i < 32; ++i) {
    float d = e4m3_qdq(v[i] * inv) * s;
    if (TRANSPOSE)
      out[(size_t)(blk * 32 + i) * NDIM + row] = (bf16)d;
    else
      out[(size_t)row * NDIM + blk * 32 + i] = (bf16)d;
  }
}

__device__ inline bf16x16 cat16(bf16x8 lo, bf16x8 hi) {
  bf16x16 r;
#pragma unroll
  for (int i = 0; i < 8; ++i) { r[i] = lo[i]; r[i + 8] = hi[i]; }
  return r;
}

#if USE_TDM
// One TDM descriptor: copy a 128-row x 64-elem bf16 tile (row stride 4096 elems)
// into LDS with 16B padding every 128B (-> LSTRIDE=144 layout). Issued per-wave.
__device__ inline void tdm_load_tile(const bf16* gbase, unsigned lds_off) {
  unsigned long long ga = (unsigned long long)(__SIZE_TYPE__)gbase;
  u32x4 g0;
  g0[0] = 1u;                                           // count=1 (valid user D#)
  g0[1] = lds_off;                                      // lds_addr (bytes)
  g0[2] = (unsigned)(ga & 0xFFFFFFFFu);                 // global_addr[31:0]
  g0[3] = (unsigned)((ga >> 32) & 0x01FFFFFFu)          // global_addr[56:32]
        | (2u << 30);                                   // type = 2 ("image")
  i32x8 g1;
  g1[0] = (1 << 16)                                     // data_size = 2 bytes
        | (1 << 20)                                     // pad_enable
        | (4 << 22)                                     // pad_interval: 32 DWORDs
        | (3 << 25);                                    // pad_amount: 4 DWORDs
  g1[1] = (int)(4096u << 16);                           // tensor_dim0[15:0] @ bits 63:48
  g1[2] = (int)(4096u << 16);                           // dim0 hi=0 | tensor_dim1[15:0]
  g1[3] = (int)(64u << 16);                             // dim1 hi=0 | tile_dim0 = 64
  g1[4] = 128;                                          // tile_dim1 = 128, tile_dim2 = 0
  g1[5] = 4096;                                         // tensor_dim0_stride lo32
  g1[6] = 0;                                            // stride0 hi | stride1 lo
  g1[7] = 0;
  i32x4 gz4 = {0, 0, 0, 0};                             // groups 2/3 unused (2D)
  i32x8 gz8 = {0, 0, 0, 0, 0, 0, 0, 0};
  __builtin_amdgcn_tensor_load_to_lds(g0, g1, gz4, gz4, gz8, 0);
}
#endif

// Stage A tile: TDM (wave 0 issues one descriptor) or per-lane async fallback.
__device__ inline void stage_a(const bf16* __restrict__ gA, int kk,
                               unsigned char* lA, int t, int m0wg) {
#if USE_TDM
  if ((t >> 5) == 0)
    tdm_load_tile(gA + (size_t)m0wg * NDIM + kk,
                  (unsigned)(__SIZE_TYPE__)(AS3 unsigned char*)(void*)lA);
#else
  const int row   = t >> 1;
  const int halfk = t & 1;
  const bf16* ga = gA + (size_t)(m0wg + row) * NDIM + kk + halfk * 32;
  unsigned char* la = lA + row * LSTRIDE + halfk * 64;
#if USE_ASYNC
#pragma unroll
  for (int i = 0; i < 4; ++i)
    __builtin_amdgcn_global_load_async_to_lds_b128(
        (i32x4*)(void*)(ga + i * 8), (AS3 i32x4*)(void*)(la + i * 16), 0, 0);
#else
#pragma unroll
  for (int i = 0; i < 4; ++i)
    *(bf16x8*)(la + i * 16) = *(const bf16x8*)(ga + i * 8);
#endif
#endif
}

// Stage B tile: per-lane async copies (VMEM async engine), 4x16B per thread.
__device__ inline void stage_b(const bf16* __restrict__ gBt, int kk,
                               unsigned char* lB, int t, int n0wg) {
  const int row   = t >> 1;
  const int halfk = t & 1;
  const bf16* gb = gBt + (size_t)(n0wg + row) * NDIM + kk + halfk * 32;
  unsigned char* lb = lB + row * LSTRIDE + halfk * 64;
#if USE_ASYNC
#pragma unroll
  for (int i = 0; i < 4; ++i)
    __builtin_amdgcn_global_load_async_to_lds_b128(
        (i32x4*)(void*)(gb + i * 8), (AS3 i32x4*)(void*)(lb + i * 16), 0, 0);
#else
#pragma unroll
  for (int i = 0; i < 4; ++i)
    *(bf16x8*)(lb + i * 16) = *(const bf16x8*)(gb + i * 8);
#endif
}

// ---------------- GEMM: C = A * B^T(N-major), LDS double-buffered, bf16 WMMA -----
template <bool REQUANT>
__global__ void __launch_bounds__(256)
mx_gemm_kernel(const bf16* __restrict__ A, const bf16* __restrict__ Bt,
               void* __restrict__ outp) {
  __shared__ __align__(16) unsigned char sA[2][128 * LSTRIDE];
  __shared__ __align__(16) unsigned char sB[2][128 * LSTRIDE];

  const int t    = threadIdx.x;
  const int lane = t & 31;
  const int w    = t >> 5;
  const int half = lane >> 4;
  const int lrow = lane & 15;
  const int m0wg = blockIdx.y * 128;
  const int n0wg = blockIdx.x * 128;
  const int am   = (w >> 2) * 64;
  const int bn   = (w & 3) * 32;

  f32x8 acc[4][2] = {};

  stage_a(A, 0, sA[0], t, m0wg);
  stage_b(Bt, 0, sB[0], t, n0wg);

  for (int kk = 0; kk < NDIM; kk += KSTEP) {
    const int cur = (kk >> 6) & 1;
    if (kk + KSTEP < NDIM) {
      stage_a(A, kk + KSTEP, sA[cur ^ 1], t, m0wg);
      stage_b(Bt, kk + KSTEP, sB[cur ^ 1], t, n0wg);
#if USE_TDM
      __builtin_amdgcn_s_wait_tensorcnt(1);            // current A done, next in flight
#endif
#if USE_ASYNC
      __builtin_amdgcn_s_wait_asynccnt(USE_TDM ? 4 : 8);
#endif
    } else {
#if USE_TDM
      __builtin_amdgcn_s_wait_tensorcnt(0);
#endif
#if USE_ASYNC
      __builtin_amdgcn_s_wait_asynccnt(0);
#endif
    }
    __syncthreads();

#pragma unroll
    for (int s = 0; s < 2; ++s) {
      bf16x16 afrag[4], bfrag[2];
#pragma unroll
      for (int i = 0; i < 4; ++i) {
        const unsigned char* p =
            sA[cur] + (am + i * 16 + lrow) * LSTRIDE + s * 64 + half * 16;
        afrag[i] = cat16(*(const bf16x8*)p, *(const bf16x8*)(p + 32));
      }
#pragma unroll
      for (int j = 0; j < 2; ++j) {
        const unsigned char* p =
            sB[cur] + (bn + j * 16 + lrow) * LSTRIDE + s * 64 + half * 16;
        bfrag[j] = cat16(*(const bf16x8*)p, *(const bf16x8*)(p + 32));
      }
#pragma unroll
      for (int i = 0; i < 4; ++i)
#pragma unroll
        for (int j = 0; j < 2; ++j)
          acc[i][j] = __builtin_amdgcn_wmma_f32_16x16x32_bf16(
              false, afrag[i], false, bfrag[j], (short)0, acc[i][j], false, false);
    }
    __syncthreads();
  }

  // Epilogue. C/D layout: VGPR r, lane -> row = r + 8*half, col = lrow.
#pragma unroll
  for (int i = 0; i < 4; ++i) {
#pragma unroll
    for (int r = 0; r < 8; ++r) {
      float v0 = acc[i][0][r];
      float v1 = acc[i][1][r];
      const int row = m0wg + am + i * 16 + half * 8 + r;
      const int n0  = n0wg + bn;
      if constexpr (REQUANT) {
        float amx = fmaxf(__builtin_fabsf(v0), __builtin_fabsf(v1));
        XORMAX(amx, 1); XORMAX(amx, 2); XORMAX(amx, 4); XORMAX(amx, 8);
        float s, inv;
        mx_scale(amx, s, inv);
        bf16* C = (bf16*)outp;
        C[(size_t)row * NDIM + n0 + lrow]      = (bf16)(e4m3_qdq(v0 * inv) * s);
        C[(size_t)row * NDIM + n0 + 16 + lrow] = (bf16)(e4m3_qdq(v1 * inv) * s);
      } else {
        float* O = (float*)outp;
        O[(size_t)row * NDIM + n0 + lrow]      = v0;
        O[(size_t)row * NDIM + n0 + 16 + lrow] = v1;
      }
    }
  }
}

extern "C" void kernel_launch(void* const* d_in, const int* in_sizes, int n_in,
                              void* d_out, int out_size, void* d_ws, size_t ws_size,
                              hipStream_t stream) {
  (void)in_sizes; (void)n_in; (void)out_size; (void)ws_size;
  const float* x  = (const float*)d_in[0];
  const float* m1 = (const float*)d_in[1];
  const float* m2 = (const float*)d_in[2];
  float* out = (float*)d_out;

  const size_t MAT = (size_t)NDIM * NDIM;
  bf16* Aq  = (bf16*)d_ws;
  bf16* B1t = Aq + MAT;
  bf16* B2t = B1t + MAT;
  bf16* Cq  = B2t + MAT;

  const int qblocks = (NDIM * (NDIM / 32)) / 256;
  mx_qdq_kernel<false><<<qblocks, 256, 0, stream>>>(x,  Aq);
  mx_qdq_kernel<true ><<<qblocks, 256, 0, stream>>>(m1, B1t);
  mx_qdq_kernel<true ><<<qblocks, 256, 0, stream>>>(m2, B2t);

  dim3 ggrid(NDIM / 128, NDIM / 128);
  mx_gemm_kernel<true ><<<ggrid, 256, 0, stream>>>(Aq, B1t, (void*)Cq);
  mx_gemm_kernel<false><<<ggrid, 256, 0, stream>>>(Cq, B2t, (void*)out);
}
